// EncoderLayer_54589034332549
// MI455X (gfx1250) — matrix operand order repaired and checked by reading, not compiled, not run on hardware
//
#include <hip/hip_runtime.h>

// ---------------------------------------------------------------------------
// CDNA5 / gfx1250 transformer encoder layer.
// B=2, S=2048, H=1024, NH=16, DH=64, FF=4096. fp32 I/O, f16 WMMA compute.
// Tile staging via GLOBAL_LOAD_ASYNC_TO_LDS_B128 (ASYNCcnt) + double buffering.
// K is produced d-major so the attention B-operand reads are contiguous b128.
// Mask is bit-packed (wave32 __ballot) -> 32x less mask traffic, L2-resident.
// ---------------------------------------------------------------------------

typedef __attribute__((ext_vector_type(16))) _Float16 v16h;
typedef __attribute__((ext_vector_type(8)))  float    v8f;

constexpr int Hd  = 1024;
constexpr int NHd = 16;
constexpr int DHd = 64;
constexpr int FFd = 4096;
constexpr int Bd  = 2;
constexpr int Sd  = 2048;
constexpr int Md  = Bd * Sd;   // 4096 rows
constexpr int SW  = Sd / 32;   // mask words per row

__device__ __forceinline__ v8f wmma_f16(v16h a, v16h b, v8f c) {
  return __builtin_amdgcn_wmma_f32_16x16x32_f16(false, a, false, b, (short)0, c, false, false);
}

// LDS aperture: flat addr[31:0] == LDS byte offset (ISA §10.2 aperture table).
__device__ __forceinline__ unsigned lds_off(const void* p) {
  return (unsigned)(unsigned long long)p;
}

// Async global->LDS 16B copy, GVS addressing (saddr base + 32-bit voffset).
// Tracked by ASYNCcnt; pairs with s_wait_asynccnt + workgroup barrier.
__device__ __forceinline__ void async_ld_b128(unsigned lds, unsigned voff,
                                              unsigned long long sbase) {
  asm volatile("global_load_async_to_lds_b128 %0, %1, %2"
               :: "v"(lds), "v"(voff), "s"(sbase) : "memory");
}
__device__ __forceinline__ void wait_async0() {
  asm volatile("s_wait_asynccnt 0" ::: "memory");
}

// ---------------------------------------------------------------------------
// Mask bit-pack: int32 [B,S,S] -> bit array [B,S,S/32] via wave32 ballot.
// ---------------------------------------------------------------------------
__global__ __launch_bounds__(256) void maskpack_kernel(const int* __restrict__ mask,
                                                       unsigned* __restrict__ bits) {
  const size_t idx = (size_t)blockIdx.x * 256 + threadIdx.x;
  const unsigned long long bal = __ballot(mask[idx] != 0);
  if ((threadIdx.x & 31) == 0) bits[idx >> 5] = (unsigned)bal;
}

// ---------------------------------------------------------------------------
// Weight convert: fp32 [O,I] (torch layout) -> f16 [I,O]; LDS-tiled transpose.
// ---------------------------------------------------------------------------
__global__ __launch_bounds__(256) void wconv_kernel(const float* __restrict__ w,
                                                    _Float16* __restrict__ wt,
                                                    int O, int I) {
  __shared__ _Float16 tile[32][33];
  const int tx = threadIdx.x & 31, ty = threadIdx.x >> 5;   // 32 x 8
  const int ob = blockIdx.x * 32, ib = blockIdx.y * 32;
#pragma unroll
  for (int s = 0; s < 32; s += 8)
    tile[ty + s][tx] = (_Float16)w[(size_t)(ob + ty + s) * I + ib + tx];
  __syncthreads();
#pragma unroll
  for (int s = 0; s < 32; s += 8)
    wt[(size_t)(ib + ty + s) * O + ob + tx] = tile[tx][ty + s];
}

// ---------------------------------------------------------------------------
// LayerNorm (fp32 in -> f16 out). One row (H=1024) per 256-thread block.
// ---------------------------------------------------------------------------
__global__ __launch_bounds__(256) void ln_f16_kernel(const float* __restrict__ x,
                                                     const float* __restrict__ w,
                                                     const float* __restrict__ bta,
                                                     _Float16* __restrict__ out) {
  const int row = blockIdx.x;
  const float* xr = x + (size_t)row * Hd;
  float s = 0.f, s2 = 0.f;
  for (int i = threadIdx.x; i < Hd; i += 256) {
    float v = xr[i];
    s += v; s2 += v * v;
  }
#pragma unroll
  for (int off = 16; off > 0; off >>= 1) {
    s  += __shfl_xor(s,  off, 32);
    s2 += __shfl_xor(s2, off, 32);
  }
  __shared__ float red0[8], red1[8];
  if ((threadIdx.x & 31) == 0) {
    red0[threadIdx.x >> 5] = s;
    red1[threadIdx.x >> 5] = s2;
  }
  __syncthreads();
  s = 0.f; s2 = 0.f;
#pragma unroll
  for (int i = 0; i < 8; ++i) { s += red0[i]; s2 += red1[i]; }
  const float mu  = s  * (1.0f / Hd);
  const float var = s2 * (1.0f / Hd) - mu * mu;
  const float inv = rsqrtf(var + 1e-6f);
  for (int i = threadIdx.x; i < Hd; i += 256)
    out[(size_t)row * Hd + i] = (_Float16)(((xr[i] - mu) * inv) * w[i] + bta[i]);
}

// ---------------------------------------------------------------------------
// Tiled WMMA GEMM:  C[M,N] = A_f16[M,K] @ Bt_f16[K,N] + bias, fused epilogue.
//   MODE 0: f16 store in head-major layout [(b,h),s,d], value scaled (Q/V)
//   MODE 1: relu -> f16 row-major (FFN1)
//   MODE 2: fp32 residual add -> outf (O proj, FFN2)
//   MODE 3: f16 store in transposed head-major layout [(b,h),d,s] (K proj)
// Block tile 128x128x32, 8 waves, wave tile 32x64. Double-buffered async LDS.
// ---------------------------------------------------------------------------
template <int MODE>
__global__ __launch_bounds__(256) void gemm_kernel(
    const _Float16* __restrict__ A, const _Float16* __restrict__ Bt,
    const float* __restrict__ bias, int N, int K, float scale,
    _Float16* __restrict__ outh, const float* __restrict__ resid,
    float* __restrict__ outf) {
  __shared__ _Float16 As[2][128][40];   // +8 pad: conflict-free A-frag reads
  __shared__ _Float16 Bs[2][32][136];   // +8 pad: conflict-free B-frag reads

  const int t    = threadIdx.x;
  const int lane = t & 31;
  const int wave = t >> 5;
  const int g = lane >> 4;
  const int r = lane & 15;
  const int wm = wave >> 1;             // 0..3 (M)
  const int wn = wave & 1;              // 0..1 (N)
  const int blockN = blockIdx.x * 128;
  const int blockM = blockIdx.y * 128;

  // per-thread async-copy geometry (loop-invariant voffsets, bytes)
  const int arow = t >> 1, acol = (t & 1) * 16;   // A tile: 128 x 32 halfs
  const int brow = t >> 3, bcol = (t & 7) * 16;   // B tile:  32 x 128 halfs
  const unsigned voffA = (unsigned)(((size_t)(blockM + arow) * K + acol) * 2);
  const unsigned voffB = (unsigned)(((size_t)brow * N + blockN + bcol) * 2);
  unsigned ldsA[2], ldsB[2];
#pragma unroll
  for (int bf = 0; bf < 2; ++bf) {
    ldsA[bf] = lds_off(&As[bf][arow][acol]);
    ldsB[bf] = lds_off(&Bs[bf][brow][bcol]);
  }
  const unsigned long long baseA = (unsigned long long)A;
  const unsigned long long baseB = (unsigned long long)Bt;
  const unsigned long long bstrideK = (unsigned long long)N * 2;  // bytes per K row

  v8f acc[2][4] = {};

  // prologue: stage tile kt=0 into buffer 0
  async_ld_b128(ldsA[0], voffA, baseA);
  async_ld_b128(ldsA[0] + 16, voffA + 16, baseA);
  async_ld_b128(ldsB[0], voffB, baseB);
  async_ld_b128(ldsB[0] + 16, voffB + 16, baseB);

  int buf = 0;
  for (int kt = 0; kt < K; kt += 32) {
    wait_async0();        // own async writes done
    __syncthreads();      // everyone's async writes visible
    const int ktn = kt + 32;
    if (ktn < K) {        // stage next tile into the other buffer
      const unsigned long long sA = baseA + (unsigned long long)ktn * 2;
      const unsigned long long sB = baseB + (unsigned long long)ktn * bstrideK;
      async_ld_b128(ldsA[buf ^ 1], voffA, sA);
      async_ld_b128(ldsA[buf ^ 1] + 16, voffA + 16, sA);
      async_ld_b128(ldsB[buf ^ 1], voffB, sB);
      async_ld_b128(ldsB[buf ^ 1] + 16, voffB + 16, sB);
    }

    v16h af[2], bfv[4];
#pragma unroll
    for (int mi = 0; mi < 2; ++mi) {
      const int mb = wm * 32 + mi * 16 + r;
#pragma unroll
      for (int i = 0; i < 8; ++i) {
        const int k0 = (i >> 2) * 16 + g * 8 + (i & 3) * 2;   // ISA 16-bit A 16x32
        af[mi][2 * i]     = As[buf][mb][k0];
        af[mi][2 * i + 1] = As[buf][mb][k0 + 1];
      }
    }
    const int kk = g * 16 + r;
#pragma unroll
    for (int ni = 0; ni < 4; ++ni) {
      const int nb = wn * 64 + ni * 16;
#pragma unroll
      for (int i = 0; i < 8; ++i) {
        const int k0 = (i >> 2) * 16 + g * 8 + (i & 3) * 2;
        bfv[ni][2 * i]     = Bs[buf][k0][nb + r];
        bfv[ni][2 * i + 1] = Bs[buf][k0 + 1][nb + r];
      }
    }
#pragma unroll
    for (int mi = 0; mi < 2; ++mi)
#pragma unroll
      for (int ni = 0; ni < 4; ++ni)
        acc[mi][ni] = wmma_f16(af[mi], bfv[ni], acc[mi][ni]);
    buf ^= 1;
  }

  // Epilogue. C/D layout: VGPR j, lane (g,r) -> M = j + 8g, N = r.
#pragma unroll
  for (int mi = 0; mi < 2; ++mi)
#pragma unroll
    for (int ni = 0; ni < 4; ++ni)
#pragma unroll
      for (int j = 0; j < 8; ++j) {
        const int row = blockM + wm * 32 + mi * 16 + g * 8 + j;
        const int col = blockN + wn * 64 + ni * 16 + r;
        const float v = acc[mi][ni][j] + bias[col];
        if (MODE == 0) {
          const int bb = row >> 11, ss = row & (Sd - 1);
          const int hh = col >> 6,  dd = col & (DHd - 1);
          outh[(((size_t)(bb * NHd + hh)) * Sd + ss) * DHd + dd] = (_Float16)(v * scale);
        } else if (MODE == 1) {
          outh[(size_t)row * N + col] = (_Float16)fmaxf(v, 0.f);
        } else if (MODE == 2) {
          const size_t idx = (size_t)row * N + col;
          outf[idx] = resid[idx] + v;
        } else {  // MODE 3: K stored d-major: [(b,h), d, s]
          const int bb = row >> 11, ss = row & (Sd - 1);
          const int hh = col >> 6,  dd = col & (DHd - 1);
          outh[(((size_t)(bb * NHd + hh)) * DHd + dd) * Sd + ss] = (_Float16)v;
        }
      }
}

// ---------------------------------------------------------------------------
// Flash attention: per (b,h) head, 128 query rows per block (16 per wave).
// Online softmax; S = Q K^T and O += P V via WMMA f16. K/V tiles staged with
// async global->LDS copies, double buffered. K arrives d-major so the score
// B-operand fragment reads are contiguous. Mask read as packed bits.
// ---------------------------------------------------------------------------
__global__ __launch_bounds__(256) void attn_kernel(
    const _Float16* __restrict__ q, const _Float16* __restrict__ k,
    const _Float16* __restrict__ v, const unsigned* __restrict__ mbits,
    _Float16* __restrict__ ctx) {
  const int bh = blockIdx.y;
  const int b  = bh >> 4;
  const int h  = bh & 15;
  const int t    = threadIdx.x;
  const int lane = t & 31;
  const int wave = t >> 5;
  const int g = lane >> 4, r = lane & 15;
  const int qbase = blockIdx.x * 128 + wave * 16;

  const _Float16* qh = q + (size_t)bh * Sd * DHd;   // [s][d]
  const _Float16* kh = k + (size_t)bh * Sd * DHd;   // [d][s]  (MODE 3 producer)
  const _Float16* vh = v + (size_t)bh * Sd * DHd;   // [s][d]
  const unsigned* mrowb = mbits + (size_t)b * Sd * SW;

  // Q fragment: 16 rows x 64 (two K=32 A-operands), kept in registers.
  v16h qf[2];
#pragma unroll
  for (int half = 0; half < 2; ++half)
#pragma unroll
    for (int i = 0; i < 8; ++i) {
      const int d = half * 32 + (i >> 2) * 16 + g * 8 + (i & 3) * 2;
      qf[half][2 * i]     = qh[(size_t)(qbase + r) * DHd + d];
      qf[half][2 * i + 1] = qh[(size_t)(qbase + r) * DHd + d + 1];
    }

  __shared__ _Float16 Kt[2][64][72];       // [d][key] tile (+pad), 2 bufs
  __shared__ _Float16 Vt[2][64][72];       // [key][d] tile (+pad), 2 bufs
  __shared__ _Float16 Ps[8][16][40];       // per-wave P staging (16 q x 32 keys)

  const int ldr = t >> 2;                  // async tile copy geometry
  const int ldc = (t & 3) * 16;
  const unsigned voffK = (unsigned)((ldr * Sd + ldc) * 2);   // K: row=d, col=s
  const unsigned voffV = (unsigned)((ldr * DHd + ldc) * 2);  // V: row=s, col=d
  unsigned ldsK[2], ldsV[2];
#pragma unroll
  for (int bf = 0; bf < 2; ++bf) {
    ldsK[bf] = lds_off(&Kt[bf][ldr][ldc]);
    ldsV[bf] = lds_off(&Vt[bf][ldr][ldc]);
  }
  const unsigned long long baseK = (unsigned long long)kh;
  const unsigned long long baseV = (unsigned long long)vh;

  float mrun[8], lrun[8];
#pragma unroll
  for (int j = 0; j < 8; ++j) { mrun[j] = -1e30f; lrun[j] = 0.f; }
  v8f o[4] = {};

  // prologue: stage kb=0 into buffer 0
  async_ld_b128(ldsK[0], voffK, baseK);
  async_ld_b128(ldsK[0] + 16, voffK + 16, baseK);
  async_ld_b128(ldsV[0], voffV, baseV);
  async_ld_b128(ldsV[0] + 16, voffV + 16, baseV);

  int buf = 0;
  for (int kb = 0; kb < Sd; kb += 64) {
    wait_async0();
    __syncthreads();
    const int kbn = kb + 64;
    if (kbn < Sd) {
      const unsigned long long offK = (unsigned long long)kbn * 2;         // +64 keys
      const unsigned long long offV = (unsigned long long)kbn * DHd * 2;   // +64 rows
      async_ld_b128(ldsK[buf ^ 1], voffK, baseK + offK);
      async_ld_b128(ldsK[buf ^ 1] + 16, voffK + 16, baseK + offK);
      async_ld_b128(ldsV[buf ^ 1], voffV, baseV + offV);
      async_ld_b128(ldsV[buf ^ 1] + 16, voffV + 16, baseV + offV);
    }

    // mask words for this 64-key tile: one dword covers 32 keys of one row
    unsigned mw[2][8];
    const int wbase = kb >> 5;
#pragma unroll
    for (int c = 0; c < 2; ++c)
#pragma unroll
      for (int j = 0; j < 8; ++j)
        mw[c][j] = mrowb[(size_t)(qbase + 8 * g + j) * SW + wbase + c];

#pragma unroll
    for (int c = 0; c < 2; ++c) {          // two 32-key chunks per tile
      float pv[2][8];
#pragma unroll
      for (int n2 = 0; n2 < 2; ++n2) {
        const int keyoff = c * 32 + n2 * 16;
        v8f sf = {};
        v16h kf0, kf1;                      // B operand: B(d, key) = Kt[d][key]
        const int dd0 = g * 16 + r;
#pragma unroll
        for (int i = 0; i < 8; ++i) {
          const int k0 = (i >> 2) * 16 + g * 8 + (i & 3) * 2;
          kf0[2 * i]     = Kt[buf][k0][keyoff + r];
          kf0[2 * i + 1] = Kt[buf][k0 + 1][keyoff + r];
          kf1[2 * i]     = Kt[buf][k0 + 32][keyoff + r];
          kf1[2 * i + 1] = Kt[buf][k0 + 33][keyoff + r];
        }
        sf = wmma_f16(qf[0], kf0, sf);
        sf = wmma_f16(qf[1], kf1, sf);
        const int bit = n2 * 16 + r;
#pragma unroll
        for (int j = 0; j < 8; ++j) {
          float xsc = sf[j];                // Q pre-scaled by DH^-0.5
          if ((mw[c][j] >> bit) & 1u) xsc = -1e30f;
          pv[n2][j] = xsc;
        }
      }
      // ---- one online-softmax update over the 32-key chunk ----
      float rmax[8];
#pragma unroll
      for (int j = 0; j < 8; ++j) rmax[j] = fmaxf(pv[0][j], pv[1][j]);
#pragma unroll
      for (int off = 8; off > 0; off >>= 1)
#pragma unroll
        for (int j = 0; j < 8; ++j) rmax[j] = fmaxf(rmax[j], __shfl_xor(rmax[j], off, 16));
      float rsum[8];
#pragma unroll
      for (int j = 0; j < 8; ++j) {
        const float newm = fmaxf(mrun[j], rmax[j]);
        const float corr = __expf(mrun[j] - newm);
        mrun[j] = newm;
        lrun[j] *= corr;
#pragma unroll
        for (int dt = 0; dt < 4; ++dt) o[dt][j] *= corr;
        rsum[j] = 0.f;
      }
#pragma unroll
      for (int n2 = 0; n2 < 2; ++n2)
#pragma unroll
        for (int j = 0; j < 8; ++j) {
          const float pe = __expf(pv[n2][j] - mrun[j]);
          rsum[j] += pe;
          Ps[wave][8 * g + j][n2 * 16 + r] = (_Float16)pe;   // C-layout -> LDS
        }
#pragma unroll
      for (int off = 8; off > 0; off >>= 1)
#pragma unroll
        for (int j = 0; j < 8; ++j) rsum[j] += __shfl_xor(rsum[j], off, 16);
#pragma unroll
      for (int j = 0; j < 8; ++j) lrun[j] += rsum[j];

      // ---- reload P in A-fragment layout, accumulate O += P V ----
      v16h pf;
#pragma unroll
      for (int i = 0; i < 8; ++i) {
        const int k0 = (i >> 2) * 16 + g * 8 + (i & 3) * 2;
        pf[2 * i]     = Ps[wave][r][k0];
        pf[2 * i + 1] = Ps[wave][r][k0 + 1];
      }
      const int kk = c * 32 + g * 16 + r;
#pragma unroll
      for (int dt = 0; dt < 4; ++dt) {
        v16h vf;                            // B operand: B(key, d) = Vt[key][d]
#pragma unroll
        for (int i = 0; i < 8; ++i) {
          const int k0 = c * 32 + (i >> 2) * 16 + g * 8 + (i & 3) * 2;
          vf[2 * i]     = Vt[buf][k0][dt * 16 + r];
          vf[2 * i + 1] = Vt[buf][k0 + 1][dt * 16 + r];
        }
        o[dt] = wmma_f16(pf, vf, o[dt]);
      }
    }
    buf ^= 1;
  }

  // normalize and write ctx back in [(b,s),(h,d)] row-major layout
#pragma unroll
  for (int dt = 0; dt < 4; ++dt)
#pragma unroll
    for (int j = 0; j < 8; ++j) {
      const float val = o[dt][j] / lrun[j];
      const int ss = qbase + 8 * g + j;
      const size_t m = (size_t)b * Sd + ss;
      const int col = h * DHd + dt * 16 + r;
      ctx[m * Hd + col] = (_Float16)val;
    }
}

// ---------------------------------------------------------------------------
// Orchestration
// ---------------------------------------------------------------------------
extern "C" void kernel_launch(void* const* d_in, const int* in_sizes, int n_in,
                              void* d_out, int out_size, void* d_ws, size_t ws_size,
                              hipStream_t stream) {
  (void)in_sizes; (void)n_in; (void)out_size; (void)ws_size;
  const float* x    = (const float*)d_in[0];
  const int*   mask = (const int*)  d_in[1];
  const float* ln0w = (const float*)d_in[2];
  const float* ln0b = (const float*)d_in[3];
  const float* ln1w = (const float*)d_in[4];
  const float* ln1b = (const float*)d_in[5];
  const float* wq = (const float*)d_in[6];  const float* bq = (const float*)d_in[7];
  const float* wk = (const float*)d_in[8];  const float* bk = (const float*)d_in[9];
  const float* wv = (const float*)d_in[10]; const float* bv = (const float*)d_in[11];
  const float* wo = (const float*)d_in[12]; const float* bo = (const float*)d_in[13];
  const float* w1 = (const float*)d_in[14]; const float* b1 = (const float*)d_in[15];
  const float* w2 = (const float*)d_in[16]; const float* b2 = (const float*)d_in[17];
  float* out = (float*)d_out;

  char* p = (char*)d_ws;
  auto take = [&](size_t bytes) -> char* {
    char* q = p;
    p += (bytes + 255) & ~(size_t)255;
    return q;
  };
  _Float16* wqT = (_Float16*)take(sizeof(_Float16) * Hd * Hd);
  _Float16* wkT = (_Float16*)take(sizeof(_Float16) * Hd * Hd);
  _Float16* wvT = (_Float16*)take(sizeof(_Float16) * Hd * Hd);
  _Float16* woT = (_Float16*)take(sizeof(_Float16) * Hd * Hd);
  _Float16* w1T = (_Float16*)take(sizeof(_Float16) * Hd * FFd);
  _Float16* w2T = (_Float16*)take(sizeof(_Float16) * FFd * Hd);
  _Float16* n0  = (_Float16*)take(sizeof(_Float16) * Md * Hd);
  _Float16* qhB = (_Float16*)take(sizeof(_Float16) * Md * Hd);
  _Float16* khB = (_Float16*)take(sizeof(_Float16) * Md * Hd);
  _Float16* vhB = (_Float16*)take(sizeof(_Float16) * Md * Hd);
  _Float16* ctxh = (_Float16*)take(sizeof(_Float16) * Md * Hd);
  float*    x1  = (float*)take(sizeof(float) * Md * Hd);
  unsigned* mb  = (unsigned*)take(sizeof(unsigned) * (size_t)Bd * Sd * SW);
  _Float16* n1  = n0;                 // reuse: n0 dead after V projection
  _Float16* yh  = qhB;                // reuse: qkv+ctx (32MB contiguous) dead after O proj

  // 0) mask bit-pack (1 MB, stays L2-resident across all 32 heads)
  maskpack_kernel<<<(size_t)Bd * Sd * Sd / 256, 256, 0, stream>>>(mask, mb);

  // 1) weight convert/transpose (f16, K-major B operands)
  wconv_kernel<<<dim3(Hd / 32, Hd / 32),  256, 0, stream>>>(wq, wqT, Hd, Hd);
  wconv_kernel<<<dim3(Hd / 32, Hd / 32),  256, 0, stream>>>(wk, wkT, Hd, Hd);
  wconv_kernel<<<dim3(Hd / 32, Hd / 32),  256, 0, stream>>>(wv, wvT, Hd, Hd);
  wconv_kernel<<<dim3(Hd / 32, Hd / 32),  256, 0, stream>>>(wo, woT, Hd, Hd);
  wconv_kernel<<<dim3(FFd / 32, Hd / 32), 256, 0, stream>>>(w1, w1T, FFd, Hd);
  wconv_kernel<<<dim3(Hd / 32, FFd / 32), 256, 0, stream>>>(w2, w2T, Hd, FFd);

  // 2) LN0
  ln_f16_kernel<<<Md, 256, 0, stream>>>(x, ln0w, ln0b, n0);

  // 3) Q/K/V projections: Q fused with DH^-0.5 scale; K stored d-major (MODE 3)
  const dim3 gH(Hd / 128, Md / 128);
  gemm_kernel<0><<<gH, 256, 0, stream>>>(n0, wqT, bq, Hd, Hd, 0.125f, qhB, nullptr, nullptr);
  gemm_kernel<3><<<gH, 256, 0, stream>>>(n0, wkT, bk, Hd, Hd, 1.0f,   khB, nullptr, nullptr);
  gemm_kernel<0><<<gH, 256, 0, stream>>>(n0, wvT, bv, Hd, Hd, 1.0f,   vhB, nullptr, nullptr);

  // 4) fused flash attention
  attn_kernel<<<dim3(Sd / 128, Bd * NHd), 256, 0, stream>>>(qhB, khB, vhB, mb, ctxh);

  // 5) output projection + residual (fp32)
  gemm_kernel<2><<<gH, 256, 0, stream>>>(ctxh, woT, bo, Hd, Hd, 1.0f, nullptr, x, x1);

  // 6) LN1
  ln_f16_kernel<<<Md, 256, 0, stream>>>(x1, ln1w, ln1b, n1);

  // 7) FFN1 (relu, f16 out) and FFN2 (+residual, fp32 out -> d_out)
  gemm_kernel<1><<<dim3(FFd / 128, Md / 128), 256, 0, stream>>>(n1, w1T, b1, FFd, Hd, 1.0f, yh, nullptr, nullptr);
  gemm_kernel<2><<<gH, 256, 0, stream>>>(yh, w2T, b2, Hd, FFd, 1.0f, nullptr, x1, out);
}
